// RelationalGAT_22995254903535
// MI455X (gfx1250) — matrix operand order repaired and checked by reading, not compiled, not run on hardware
//
#include <hip/hip_runtime.h>
#include <stdint.h>

#define NEG_SLOPE 0.2f

typedef __attribute__((ext_vector_type(16))) __bf16 v16bf;
typedef __attribute__((ext_vector_type(8)))  __bf16 v8bf;
typedef __attribute__((ext_vector_type(8)))  float  v8f;

__device__ __forceinline__ unsigned short f2bf(float f) {
  unsigned int u = __float_as_uint(f);
  u += 0x7fffu + ((u >> 16) & 1u);            // round-to-nearest-even
  return (unsigned short)(u >> 16);
}
__device__ __forceinline__ int f2ord(float f) {   // order-preserving float->int
  int i = __float_as_int(f);
  return (i >= 0) ? i : (i ^ 0x7fffffff);
}
__device__ __forceinline__ float ord2f(int i) {
  int j = (i >= 0) ? i : (i ^ 0x7fffffff);
  return __int_as_float(j);
}
__device__ __forceinline__ float lrelu(float x) { return x > 0.f ? x : NEG_SLOPE * x; }

// ---------------- fill helpers ----------------
__global__ void rgat_fill_f32(float* __restrict__ p, float v, long long n) {
  long long i = (long long)blockIdx.x * blockDim.x + threadIdx.x;
  if (i < n) p[i] = v;
}
__global__ void rgat_fill_i32(int* __restrict__ p, int v, long long n) {
  long long i = (long long)blockIdx.x * blockDim.x + threadIdx.x;
  if (i < n) p[i] = v;
}

// ---------------- prep ----------------
// copy node features to working f32 buffer + bf16 mirror
__global__ void rgat_init_x(const float* __restrict__ xin, float* __restrict__ xcur,
                            unsigned short* __restrict__ xb, int n128) {
  int i = blockIdx.x * blockDim.x + threadIdx.x;
  if (i >= n128) return;
  float v = xin[i];
  xcur[i] = v;
  xb[i] = f2bf(v);
}

// Pack a 128x128 f32 weight (K rows x N cols, row-major) into the WMMA B-fragment
// lane layout for 16x16x32 bf16: Wp[((ntile*4 + kc)*32 + lane)*16 + j]
// lane<16  -> N = ntile*16 + lane,    K = kc*32 + (j<8 ? j : 16 + j-8)      (j = 0..15)
// lane>=16 -> N = ntile*16 + lane-16, K = kc*32 + (j<8 ? 8 + j : 24 + j-8)
__global__ void rgat_pack_w(const float* __restrict__ W, unsigned short* __restrict__ Wp) {
  int i = blockIdx.x * blockDim.x + threadIdx.x;  // 8*4*32*16 = 16384
  if (i >= 16384) return;
  int j    = i & 15;
  int lane = (i >> 4) & 31;
  int kc   = (i >> 9) & 3;
  int nt   = i >> 11;
  int n = nt * 16 + (lane & 15);
  int k = kc * 32 + (lane >> 4) * 8 + (j < 8 ? j : 16 + (j - 8));
  Wp[i] = f2bf(W[k * 128 + n]);
}

// rel_ee[r, j] = sum_k rel_emb[r,k] * We[k,j]   (R x 64 @ 64 x 128, tiny)
__global__ void rgat_rel_gemm(const float* __restrict__ rel, const float* __restrict__ We,
                              float* __restrict__ relee, int R) {
  int i = blockIdx.x * blockDim.x + threadIdx.x;
  if (i >= R * 128) return;
  int r = i >> 7, j = i & 127;
  float s = 0.f;
#pragma unroll 8
  for (int k = 0; k < 64; ++k) s += rel[r * 64 + k] * We[k * 128 + j];
  relee[i] = s;
}

__global__ void rgat_deg(const int* __restrict__ dst, float* __restrict__ deg, int E) {
  int i = blockIdx.x * blockDim.x + threadIdx.x;
  if (i < E) atomicAdd(&deg[dst[i]], 1.0f);
}

// slsum[dst, d] += rel_ee[etype, d]   (one thread per (edge, dim))
__global__ void rgat_sl_accum(const float* __restrict__ relee, const int* __restrict__ dst,
                              const int* __restrict__ etype, float* __restrict__ slsum, int E) {
  int i = blockIdx.x * blockDim.x + threadIdx.x;
  if (i >= E * 128) return;
  int e = i >> 7, d7 = i & 127;
  atomicAdd(&slsum[(size_t)dst[e] * 128 + d7], relee[(size_t)etype[e] * 128 + d7]);
}

__global__ void rgat_sl_div(float* __restrict__ sl, const float* __restrict__ deg, int n128) {
  int i = blockIdx.x * blockDim.x + threadIdx.x;
  if (i >= n128) return;
  sl[i] = sl[i] / fmaxf(deg[i >> 7], 1.0f);
}

// ---------------- WMMA GEMM: Y[M,128] = bf16(X)[M,128] @ bf16(W)[128,128] + bias ----------------
// 4 waves/block, each wave owns one 16(M) x 16(N) tile; K = 128 in four x32 chunks.
__global__ void rgat_gemm_bf16(const unsigned short* __restrict__ Xb,
                               const unsigned short* __restrict__ Wp,
                               const float* __restrict__ bias,
                               float* __restrict__ Y, int M) {
  const int lane  = threadIdx.x & 31;
  const int wave  = threadIdx.x >> 5;
  const int row0  = blockIdx.x * 16;
  const int ntile = blockIdx.y * 4 + wave;     // 0..7
  const int col0  = ntile * 16;
  const int khalf = lane >> 4;
  int mrow = row0 + (lane & 15);
  if (mrow >= M) mrow = M - 1;                 // clamp (EXEC must stay all-ones for WMMA)

  v8f acc = {};
#pragma unroll
  for (int kc = 0; kc < 4; ++kc) {
    // A fragment (16x32 bf16): lane<16 holds K[0..7]+K[16..23] of row m; lane>=16 holds K[8..15]+K[24..31]
    const unsigned short* ap = Xb + (size_t)mrow * 128 + kc * 32 + khalf * 8;
    v8bf alo = *(const v8bf*)(ap);
    v8bf ahi = *(const v8bf*)(ap + 16);
    v16bf A = __builtin_shufflevector(alo, ahi, 0,1,2,3,4,5,6,7,8,9,10,11,12,13,14,15);
    // B fragment (32x16 bf16) pre-swizzled: 16 contiguous bf16 per lane
    const unsigned short* bp = Wp + ((size_t)(ntile * 4 + kc) * 32 + lane) * 16;
    v8bf blo = *(const v8bf*)(bp);
    v8bf bhi = *(const v8bf*)(bp + 8);
    v16bf B = __builtin_shufflevector(blo, bhi, 0,1,2,3,4,5,6,7,8,9,10,11,12,13,14,15);
    acc = __builtin_amdgcn_wmma_f32_16x16x32_bf16(false, A, false, B, (short)0, acc, false, false);
  }
  // C/D layout: VGPR r -> row = row0 + r + 8*khalf, col = col0 + (lane&15)
  const int col = col0 + (lane & 15);
  const float bv = bias[col];
#pragma unroll
  for (int r = 0; r < 8; ++r) {
    int row = row0 + r + 8 * khalf;
    if (row < M) Y[(size_t)row * 128 + col] = acc[r] + bv;
  }
}

// ---------------- edge kernels (one wave per edge; lanes stride the 128 dims x4) ----------------
template <int H>
__global__ void rgat_edge_logits(const float* __restrict__ xl, const float* __restrict__ xr,
                                 const float* __restrict__ relee, const float* __restrict__ slee,
                                 const int* __restrict__ src, const int* __restrict__ dst,
                                 const int* __restrict__ etype, const float* __restrict__ att,
                                 float* __restrict__ logits, int* __restrict__ mx,
                                 int E, int N) {
  const int wid  = (int)(((size_t)blockIdx.x * blockDim.x + threadIdx.x) >> 5);
  const int lane = threadIdx.x & 31;
  if (wid >= E + N) return;
  int s, d;
  const float* ee;
  if (wid < E) { s = src[wid]; d = dst[wid]; ee = relee + (size_t)etype[wid] * 128; }
  else         { s = wid - E; d = s;         ee = slee  + (size_t)s * 128; }
  const int dd = lane * 4;
  float4 a  = *(const float4*)(xl + (size_t)s * 128 + dd);
  float4 b  = *(const float4*)(xr + (size_t)d * 128 + dd);
  float4 e4 = *(const float4*)(ee + dd);
  float4 t4 = *(const float4*)(att + dd);
  float acc = lrelu(a.x + b.x + e4.x) * t4.x
            + lrelu(a.y + b.y + e4.y) * t4.y
            + lrelu(a.z + b.z + e4.z) * t4.z
            + lrelu(a.w + b.w + e4.w) * t4.w;
  constexpr int G = 32 / H;                       // lanes per head
#pragma unroll
  for (int off = 1; off < G; off <<= 1) acc += __shfl_xor(acc, off, 32);
  if ((lane & (G - 1)) == 0) {
    const int h = lane / G;
    logits[(size_t)wid * H + h] = acc;
    atomicMax(mx + (size_t)d * H + h, f2ord(acc));  // segment max over dst
  }
}

template <int H>
__global__ void rgat_edge_exp(const float* __restrict__ logits, const int* __restrict__ mx,
                              const int* __restrict__ dst, float* __restrict__ ex,
                              float* __restrict__ den, int E, int N) {
  int i = blockIdx.x * blockDim.x + threadIdx.x;
  if (i >= (E + N) * H) return;
  int e = i / H, h = i - e * H;
  int d = (e < E) ? dst[e] : (e - E);
  float v = __expf(logits[i] - ord2f(mx[(size_t)d * H + h]));
  ex[i] = v;
  atomicAdd(den + (size_t)d * H + h, v);
}

template <int H>
__global__ void rgat_edge_scatter(const float* __restrict__ xl, const float* __restrict__ ex,
                                  const float* __restrict__ den, const int* __restrict__ src,
                                  const int* __restrict__ dst, float* __restrict__ out,
                                  float* __restrict__ alphaOut, int E, int N) {
  const int wid  = (int)(((size_t)blockIdx.x * blockDim.x + threadIdx.x) >> 5);
  const int lane = threadIdx.x & 31;
  if (wid >= E + N) return;
  int s, d;
  if (wid < E) { s = src[wid]; d = dst[wid]; } else { s = wid - E; d = s; }
  constexpr int C = 128 / H;
  const int dd = lane * 4;
  const int h  = dd / C;                         // all 4 dims fall in one head (C % 4 == 0)
  const float alpha = ex[(size_t)wid * H + h] / den[(size_t)d * H + h];
  float4 xv = *(const float4*)(xl + (size_t)s * 128 + dd);
  float* op = out + (size_t)d * 128 + dd;
  atomicAdd(op + 0, xv.x * alpha);
  atomicAdd(op + 1, xv.y * alpha);
  atomicAdd(op + 2, xv.z * alpha);
  atomicAdd(op + 3, xv.w * alpha);
  if (H == 1 && alphaOut != nullptr && wid < E && lane == 0) alphaOut[wid] = alpha;
}

// x += elu(agg + bias); refresh bf16 mirror; optionally mirror into final output
__global__ void rgat_residual_elu(const float* __restrict__ agg, const float* __restrict__ bias,
                                  float* __restrict__ xcur, unsigned short* __restrict__ xb,
                                  float* __restrict__ xout, int n128) {
  int i = blockIdx.x * blockDim.x + threadIdx.x;
  if (i >= n128) return;
  float g = agg[i] + bias[i & 127];
  float el = g > 0.f ? g : (__expf(g) - 1.0f);
  float nx = xcur[i] + el;
  xcur[i] = nx;
  xb[i] = f2bf(nx);
  if (xout != nullptr) xout[i] = nx;
}

// ---------------- host ----------------
extern "C" void kernel_launch(void* const* d_in, const int* in_sizes, int n_in,
                              void* d_out, int out_size, void* d_ws, size_t ws_size,
                              hipStream_t stream) {
  (void)n_in; (void)out_size; (void)ws_size;
  const float* x_in  = (const float*)d_in[0];
  const int*   eidx  = (const int*)d_in[1];
  const int*   etype = (const int*)d_in[2];
  const float* rel   = (const float*)d_in[3];
  const float* Wl0 = (const float*)d_in[4];  const float* bl0  = (const float*)d_in[5];
  const float* Wr0 = (const float*)d_in[6];  const float* br0  = (const float*)d_in[7];
  const float* We0 = (const float*)d_in[8];  const float* att0 = (const float*)d_in[9];
  const float* bias0 = (const float*)d_in[10];
  const float* Wl1 = (const float*)d_in[11]; const float* bl1  = (const float*)d_in[12];
  const float* Wr1 = (const float*)d_in[13]; const float* br1  = (const float*)d_in[14];
  const float* We1 = (const float*)d_in[15]; const float* att1 = (const float*)d_in[16];
  const float* bias1 = (const float*)d_in[17];

  const int N = in_sizes[0] / 128;
  const int E = in_sizes[2];
  const int R = in_sizes[3] / 64;
  const int* src  = eidx;
  const int* dstp = eidx + E;

  char* wsp = (char*)d_ws;
  auto carve = [&](size_t bytes) -> void* {
    void* q = (void*)wsp;
    wsp += (bytes + 255) & ~(size_t)255;
    return q;
  };
  float*          xcur   = (float*)carve((size_t)N * 128 * 4);
  unsigned short* xb     = (unsigned short*)carve((size_t)N * 128 * 2);
  float*          xl     = (float*)carve((size_t)N * 128 * 4);
  float*          xr     = (float*)carve((size_t)N * 128 * 4);
  float*          ebuf   = (float*)carve((size_t)N * 128 * 4);   // sl_ee, then aggregation out
  float*          deg    = (float*)carve((size_t)N * 4);
  float*          relee0 = (float*)carve((size_t)R * 128 * 4);
  float*          relee1 = (float*)carve((size_t)R * 128 * 4);
  unsigned short* wl0p   = (unsigned short*)carve(16384 * 2);
  unsigned short* wr0p   = (unsigned short*)carve(16384 * 2);
  unsigned short* wl1p   = (unsigned short*)carve(16384 * 2);
  unsigned short* wr1p   = (unsigned short*)carve(16384 * 2);
  float*          logits = (float*)carve((size_t)(E + N) * 4 * 4);
  float*          exb    = (float*)carve((size_t)(E + N) * 4 * 4);
  int*            mx     = (int*)carve((size_t)N * 4 * 4);
  float*          den    = (float*)carve((size_t)N * 4 * 4);

  const int TB = 256;
  auto nb = [](long long n, int b) { return (unsigned)((n + b - 1) / b); };
  const long long n128  = (long long)N * 128;
  const long long waves = (long long)(E + N) * 32;

  // --- prep ---
  rgat_init_x<<<nb(n128, TB), TB, 0, stream>>>(x_in, xcur, xb, (int)n128);
  rgat_pack_w<<<nb(16384, TB), TB, 0, stream>>>(Wl0, wl0p);
  rgat_pack_w<<<nb(16384, TB), TB, 0, stream>>>(Wr0, wr0p);
  rgat_pack_w<<<nb(16384, TB), TB, 0, stream>>>(Wl1, wl1p);
  rgat_pack_w<<<nb(16384, TB), TB, 0, stream>>>(Wr1, wr1p);
  rgat_rel_gemm<<<nb((long long)R * 128, TB), TB, 0, stream>>>(rel, We0, relee0, R);
  rgat_rel_gemm<<<nb((long long)R * 128, TB), TB, 0, stream>>>(rel, We1, relee1, R);
  rgat_fill_f32<<<nb(N, TB), TB, 0, stream>>>(deg, 0.f, N);
  rgat_deg<<<nb(E, TB), TB, 0, stream>>>(dstp, deg, E);

  float* xfinal = (float*)d_out;
  float* alpha1 = (float*)d_out + (size_t)N * 128;

  dim3 gg((unsigned)((N + 15) / 16), 2), gbk(128);

  // --- layer 0 (H=4, C=32, concat) ---
  rgat_gemm_bf16<<<gg, gbk, 0, stream>>>(xb, wl0p, bl0, xl, N);
  rgat_gemm_bf16<<<gg, gbk, 0, stream>>>(xb, wr0p, br0, xr, N);
  rgat_fill_f32<<<nb(n128, TB), TB, 0, stream>>>(ebuf, 0.f, n128);
  rgat_sl_accum<<<nb((long long)E * 128, TB), TB, 0, stream>>>(relee0, dstp, etype, ebuf, E);
  rgat_sl_div<<<nb(n128, TB), TB, 0, stream>>>(ebuf, deg, (int)n128);
  rgat_fill_i32<<<nb((long long)N * 4, TB), TB, 0, stream>>>(mx, (int)0x80000000, (long long)N * 4);
  rgat_fill_f32<<<nb((long long)N * 4, TB), TB, 0, stream>>>(den, 0.f, (long long)N * 4);
  rgat_edge_logits<4><<<nb(waves, TB), TB, 0, stream>>>(xl, xr, relee0, ebuf, src, dstp, etype,
                                                        att0, logits, mx, E, N);
  rgat_edge_exp<4><<<nb((long long)(E + N) * 4, TB), TB, 0, stream>>>(logits, mx, dstp, exb, den, E, N);
  rgat_fill_f32<<<nb(n128, TB), TB, 0, stream>>>(ebuf, 0.f, n128);
  rgat_edge_scatter<4><<<nb(waves, TB), TB, 0, stream>>>(xl, exb, den, src, dstp, ebuf, nullptr, E, N);
  rgat_residual_elu<<<nb(n128, TB), TB, 0, stream>>>(ebuf, bias0, xcur, xb, nullptr, (int)n128);

  // --- layer 1 (H=1, C=128, mean == identity) ---
  rgat_gemm_bf16<<<gg, gbk, 0, stream>>>(xb, wl1p, bl1, xl, N);
  rgat_gemm_bf16<<<gg, gbk, 0, stream>>>(xb, wr1p, br1, xr, N);
  rgat_fill_f32<<<nb(n128, TB), TB, 0, stream>>>(ebuf, 0.f, n128);
  rgat_sl_accum<<<nb((long long)E * 128, TB), TB, 0, stream>>>(relee1, dstp, etype, ebuf, E);
  rgat_sl_div<<<nb(n128, TB), TB, 0, stream>>>(ebuf, deg, (int)n128);
  rgat_fill_i32<<<nb((long long)N, TB), TB, 0, stream>>>(mx, (int)0x80000000, (long long)N);
  rgat_fill_f32<<<nb((long long)N, TB), TB, 0, stream>>>(den, 0.f, (long long)N);
  rgat_edge_logits<1><<<nb(waves, TB), TB, 0, stream>>>(xl, xr, relee1, ebuf, src, dstp, etype,
                                                        att1, logits, mx, E, N);
  rgat_edge_exp<1><<<nb((long long)(E + N), TB), TB, 0, stream>>>(logits, mx, dstp, exb, den, E, N);
  rgat_fill_f32<<<nb(n128, TB), TB, 0, stream>>>(ebuf, 0.f, n128);
  rgat_edge_scatter<1><<<nb(waves, TB), TB, 0, stream>>>(xl, exb, den, src, dstp, ebuf, alpha1, E, N);
  rgat_residual_elu<<<nb(n128, TB), TB, 0, stream>>>(ebuf, bias1, xcur, xb, xfinal, (int)n128);
}
